// InstructPerceiverResampler_28759101014378
// MI455X (gfx1250) — matrix operand order repaired and checked by reading, not compiled, not run on hardware
//
#include <hip/hip_runtime.h>
#include <hip/hip_bf16.h>

typedef __attribute__((ext_vector_type(16))) __bf16 v16bf;
typedef __attribute__((ext_vector_type(8)))  float  v8f;

#define DIMC   1152
#define DIMLLM 4096
#define NTXTC  64
#define BATCHC 32
#define VTOK   729
#define NLATC  64
#define N2C    128
#define NKC    857
#define HEADSC 16
#define DHC    96
#define INNERC 1536
#define FFIC   4608
#define DEPTHC 6
#define QSCALE 0.10206207261596577f   // 96^-0.5

// ---------------------------------------------------------------------------
// helpers
// ---------------------------------------------------------------------------
__device__ __forceinline__ float gelu_exact(float x) {
    return 0.5f * x * (1.0f + erff(x * 0.70710678118654752f));
}
__device__ __forceinline__ float xmax16(float v) {
    #pragma unroll
    for (int m = 1; m < 16; m <<= 1) v = fmaxf(v, __shfl_xor(v, m, 32));
    return v;
}
__device__ __forceinline__ float xsum16(float v) {
    #pragma unroll
    for (int m = 1; m < 16; m <<= 1) v += __shfl_xor(v, m, 32);
    return v;
}
__device__ __forceinline__ float wsum32(float v) {
    #pragma unroll
    for (int m = 1; m < 32; m <<= 1) v += __shfl_xor(v, m, 32);
    return v;
}

__device__ __forceinline__ v8f wmma_bf16(v16bf a, v16bf b, v8f c) {
    return __builtin_amdgcn_wmma_f32_16x16x32_bf16(
        false, a, false, b, (short)0, c, false, false);
}

// A fragment (16x32 bf16, ISA 7.12.2): lane<16 -> M=lane, K={0..7,16..23};
// lane>=16 -> M=lane-16, K={8..15,24..31}.  base points at tile (row0,k0).
__device__ __forceinline__ v16bf load_a_frag(const __bf16* base, int ld, int lane) {
    int row = lane & 15, half = lane >> 4;
    union { uint4 q[2]; v16bf v; } u;
    u.q[0] = *(const uint4*)(base + row * ld + half * 8);
    u.q[1] = *(const uint4*)(base + row * ld + 16 + half * 8);
    return u.v;
}
// B fragment (32x16 bf16), B tile stored N-major [N][K]: lane -> N=lane&15,
// K = {0..7,16..23} (lane<16) or {8..15,24..31} (lane>=16), matching A.
__device__ __forceinline__ v16bf load_b_frag(const __bf16* base, int ld, int lane) {
    int row = lane & 15, half = lane >> 4;
    union { uint4 q[2]; v16bf v; } u;
    u.q[0] = *(const uint4*)(base + row * ld + half * 8);
    u.q[1] = *(const uint4*)(base + row * ld + 16 + half * 8);
    return u.v;
}

// ---------------------------------------------------------------------------
// weight transpose + convert:  out[l][n][k] = bf16(in[l][k][n])
// ---------------------------------------------------------------------------
__global__ __launch_bounds__(256) void transpose_bf16_kernel(
    const float* __restrict__ in, __bf16* __restrict__ out, int K, int N) {
    __shared__ float tile[32][33];
    int l = blockIdx.z;
    const float* ip = in + (size_t)l * K * N;
    __bf16* op = out + (size_t)l * K * N;
    int tx = threadIdx.x, ty = threadIdx.y;
    int n = blockIdx.x * 32 + tx;
    int k0 = blockIdx.y * 32;
    #pragma unroll
    for (int j = 0; j < 4; ++j)
        tile[ty + j * 8][tx] = ip[(size_t)(k0 + ty + j * 8) * N + n];
    __syncthreads();
    int k = k0 + tx;
    #pragma unroll
    for (int j = 0; j < 4; ++j) {
        int nn = blockIdx.x * 32 + ty + j * 8;
        op[(size_t)nn * K + k] = (__bf16)tile[tx][ty + j * 8];
    }
}

__global__ __launch_bounds__(256) void convert_bf16_kernel(
    const float* __restrict__ in, __bf16* __restrict__ out, size_t n) {
    size_t i = (size_t)blockIdx.x * 256 + threadIdx.x;
    if (i < n) out[i] = (__bf16)in[i];
}

// per-row mean/rstd of x + bf16 copy (LN fused later into kv GEMM)
__global__ __launch_bounds__(256) void row_stats_kernel(
    const float* __restrict__ in, __bf16* __restrict__ outbf,
    float2* __restrict__ stats, int cols) {
    int r = blockIdx.x;
    const float* row = in + (size_t)r * cols;
    float s = 0.f, s2 = 0.f;
    for (int c = threadIdx.x; c < cols; c += 256) {
        float v = row[c];
        s += v; s2 += v * v;
        outbf[(size_t)r * cols + c] = (__bf16)v;
    }
    s = wsum32(s); s2 = wsum32(s2);
    __shared__ float sh[2][8];
    int lane = threadIdx.x & 31, w = threadIdx.x >> 5;
    if (lane == 0) { sh[0][w] = s; sh[1][w] = s2; }
    __syncthreads();
    if (threadIdx.x == 0) {
        float a = 0.f, b = 0.f;
        #pragma unroll
        for (int i = 0; i < 8; ++i) { a += sh[0][i]; b += sh[1][i]; }
        float mu = a / cols;
        float var = b / cols - mu * mu;
        stats[r] = make_float2(mu, rsqrtf(fmaxf(var, 0.f) + 1e-5f));
    }
}

// layernorm rows of f32 input -> bf16 (or f32 for final output); input row remap
template <bool OUT_F32>
__global__ __launch_bounds__(256) void ln_kernel(
    const float* __restrict__ in, const float* __restrict__ w,
    const float* __restrict__ bsh, void* __restrict__ out, int cols,
    int rdiv, int rmul, int radd) {
    int r = blockIdx.x;
    int irow = (r / rdiv) * rmul + radd + (r % rdiv);
    const float* row = in + (size_t)irow * cols;
    float s = 0.f, s2 = 0.f;
    for (int c = threadIdx.x; c < cols; c += 256) {
        float v = row[c];
        s += v; s2 += v * v;
    }
    s = wsum32(s); s2 = wsum32(s2);
    __shared__ float sh[2][8];
    int lane = threadIdx.x & 31, wv = threadIdx.x >> 5;
    if (lane == 0) { sh[0][wv] = s; sh[1][wv] = s2; }
    __syncthreads();
    if (threadIdx.x == 0) {
        float a = 0.f, b = 0.f;
        #pragma unroll
        for (int i = 0; i < 8; ++i) { a += sh[0][i]; b += sh[1][i]; }
        sh[0][0] = a; sh[1][0] = b;
    }
    __syncthreads();
    float mu = sh[0][0] / cols;
    float var = sh[1][0] / cols - mu * mu;
    float rstd = rsqrtf(fmaxf(var, 0.f) + 1e-5f);
    for (int c = threadIdx.x; c < cols; c += 256) {
        float v = (row[c] - mu) * rstd * w[c] + bsh[c];
        if constexpr (OUT_F32) ((float*)out)[(size_t)r * cols + c] = v;
        else ((__bf16*)out)[(size_t)r * cols + c] = (__bf16)v;
    }
}

__global__ __launch_bounds__(256) void init_lat_kernel(
    const float* __restrict__ latents, float* __restrict__ lat) {
    int i = blockIdx.x * 256 + threadIdx.x;
    if (i < BATCHC * NLATC * DIMC) {
        int d = i % DIMC;
        int j = (i / DIMC) % NLATC;
        int b = i / (NLATC * DIMC);
        lat[((size_t)b * N2C + j) * DIMC + d] = latents[(size_t)j * DIMC + d];
    }
}

// ---------------------------------------------------------------------------
// generic bf16 WMMA GEMM:  C = A(MxK,row) @ Bt(NxK,N-major)^T
//   block tile 128x64x32, 8 waves (4x2), each wave 2x2 of 16x16 tiles
//   register-staged + double-buffered LDS, 1 barrier per K-tile
//   LN_A: (x-mu)*rstd*lnw[k]+lnb[k] applied while staging A
//   row remap: out_row = (r/rdiv)*rmul + radd + r%rdiv
// ---------------------------------------------------------------------------
#define GLDA 40  // padded LDS row stride (bf16 elems) -> conflict-free b128 reads

template <bool LN_A, bool BIAS, bool GELU_E, bool RESID, bool OUT_BF16>
__global__ __launch_bounds__(256) void gemm_kernel(
    const __bf16* __restrict__ A, const float2* __restrict__ stats,
    const float* __restrict__ lnw, const float* __restrict__ lnb,
    const __bf16* __restrict__ Bt, const float* __restrict__ bias,
    const float* __restrict__ resid, void* __restrict__ out,
    int M, int N, int K, int rdiv, int rmul, int radd) {
    __shared__ __bf16 As[2][128 * GLDA];
    __shared__ __bf16 Bs[2][64 * GLDA];
    int tid = threadIdx.x, lane = tid & 31, w = tid >> 5;
    int wm = w >> 1, wn = w & 1;
    int m0 = blockIdx.y * 128, n0 = blockIdx.x * 64;
    v8f acc[2][2] = {};
    int arow = tid >> 1, acol = (tid & 1) * 16;  // 16 bf16 per thread
    int brow = tid >> 2, bcol = (tid & 3) * 8;   // 8 bf16 per thread
    int gr = m0 + arow;
    bool aval = gr < M;
    int grs = aval ? gr : 0;
    float2 st = make_float2(0.f, 0.f);
    if constexpr (LN_A) { if (aval) st = stats[grs]; }

    uint4 ra0, ra1, rb;
    float rw[16], rbv[16];

    auto fetch = [&](int k0) {
        const uint4* ap = (const uint4*)(A + (size_t)grs * K + k0 + acol);
        ra0 = ap[0];
        ra1 = ap[1];
        rb = *(const uint4*)(Bt + (size_t)(n0 + brow) * K + k0 + bcol);
        if constexpr (LN_A) {
            #pragma unroll
            for (int j = 0; j < 4; ++j) {
                *(float4*)&rw[j * 4]  = *(const float4*)(lnw + k0 + acol + j * 4);
                *(float4*)&rbv[j * 4] = *(const float4*)(lnb + k0 + acol + j * 4);
            }
        }
    };
    auto stage = [&](int bufi) {
        union { uint4 q; __bf16 h[8]; } u0, u1;
        u0.q = ra0; u1.q = ra1;
        if (!aval) {
            u0.q = make_uint4(0, 0, 0, 0); u1.q = u0.q;
        } else {
            if constexpr (LN_A) {
                #pragma unroll
                for (int e = 0; e < 8; ++e) {
                    u0.h[e] = (__bf16)(((float)u0.h[e] - st.x) * st.y * rw[e] + rbv[e]);
                    u1.h[e] = (__bf16)(((float)u1.h[e] - st.x) * st.y * rw[e + 8] + rbv[e + 8]);
                }
            }
        }
        *(uint4*)(&As[bufi][arow * GLDA + acol]) = u0.q;
        *(uint4*)(&As[bufi][arow * GLDA + acol + 8]) = u1.q;
        *(uint4*)(&Bs[bufi][brow * GLDA + bcol]) = rb;
    };

    int ktiles = K >> 5;
    fetch(0);
    for (int kt = 0; kt < ktiles; ++kt) {
        int bufi = kt & 1;
        stage(bufi);
        if (kt + 1 < ktiles) fetch((kt + 1) << 5);
        __syncthreads();
        const __bf16* Ab = &As[bufi][0];
        const __bf16* Bb = &Bs[bufi][0];
        v16bf af0 = load_a_frag(Ab + (wm * 32) * GLDA, GLDA, lane);
        v16bf af1 = load_a_frag(Ab + (wm * 32 + 16) * GLDA, GLDA, lane);
        v16bf bf0 = load_b_frag(Bb + (wn * 32) * GLDA, GLDA, lane);
        v16bf bf1 = load_b_frag(Bb + (wn * 32 + 16) * GLDA, GLDA, lane);
        acc[0][0] = wmma_bf16(af0, bf0, acc[0][0]);
        acc[0][1] = wmma_bf16(af0, bf1, acc[0][1]);
        acc[1][0] = wmma_bf16(af1, bf0, acc[1][0]);
        acc[1][1] = wmma_bf16(af1, bf1, acc[1][1]);
    }
    // --- epilogue ---
    int half = lane >> 4, cl = lane & 15;
    #pragma unroll
    for (int i = 0; i < 2; ++i) {
        #pragma unroll
        for (int j = 0; j < 2; ++j) {
            int cg = n0 + wn * 32 + j * 16 + cl;
            #pragma unroll
            for (int r = 0; r < 8; ++r) {
                int rg = m0 + wm * 32 + i * 16 + r + half * 8;
                if (rg < M) {
                    float v = acc[i][j][r];
                    if constexpr (BIAS) v += bias[cg];
                    if constexpr (GELU_E) v = gelu_exact(v);
                    size_t orow = (size_t)(rg / rdiv) * rmul + radd + (rg % rdiv);
                    size_t idx = orow * (size_t)N + cg;
                    if constexpr (RESID) v += resid[idx];
                    if constexpr (OUT_BF16) ((__bf16*)out)[idx] = (__bf16)v;
                    else ((float*)out)[idx] = v;
                }
            }
        }
    }
}

// ---------------------------------------------------------------------------
// flash attention: one block per (b,h); 8 waves x 16 q-rows; keys in tiles of 32
//   q fragments loaded once from global straight into WMMA A-layout registers
//   S = q @ k^T (WMMA, K=96 -> 3 steps), scaled after the product
//   online softmax per wave (rows owned exclusively); P via LDS -> A-frag
//   O += P @ V (v transposed into LDS as N-major); k/v tiles double-buffered
// ---------------------------------------------------------------------------
#define QLD 104  // padded k LDS stride
#define PLD 40   // padded p/vt LDS stride

__global__ __launch_bounds__(256) void attn_kernel(
    const __bf16* __restrict__ q, const __bf16* __restrict__ kv,
    __bf16* __restrict__ o) {
    __shared__ __bf16 ks[2][32 * QLD];
    __shared__ __bf16 vts[2][96 * PLD];
    __shared__ __bf16 ps[8 * 16 * PLD];
    int tid = threadIdx.x, lane = tid & 31, w = tid >> 5;
    int half = lane >> 4, cl = lane & 15;
    int b = blockIdx.x >> 4, h = blockIdx.x & 15;
    const __bf16* qb = q + ((size_t)b * N2C * HEADSC + h) * DHC;
    const __bf16* kvb = kv + (size_t)b * NKC * (2 * INNERC) + h * DHC;
    int m0 = w * 16;

    // q A-fragments straight from global (row-major == A-frag addressing)
    v16bf qfrag[3];
    #pragma unroll
    for (int kk = 0; kk < 3; ++kk) {
        const __bf16* base = qb + (size_t)(m0 + cl) * INNERC + kk * 32 + half * 8;
        union { uint4 x[2]; v16bf v; } u;
        u.x[0] = *(const uint4*)(base);
        u.x[1] = *(const uint4*)(base + 16);
        qfrag[kk] = u.v;
    }

    auto load_tile = [&](int kt2, int bufi) {
        int kbase = kt2 * 32;
        for (int t = tid; t < 32 * 12; t += 256) {
            int row = t / 12, c = (t % 12) * 8;
            uint4 val = make_uint4(0, 0, 0, 0);
            if (kbase + row < NKC)
                val = *(const uint4*)(kvb + (size_t)(kbase + row) * (2 * INNERC) + c);
            *(uint4*)(&ks[bufi][row * QLD + c]) = val;
        }
        for (int t = tid; t < 32 * 96; t += 256) {
            int row = t / 96, d = t % 96;
            __bf16 val = (__bf16)0.0f;
            if (kbase + row < NKC)
                val = kvb[(size_t)(kbase + row) * (2 * INNERC) + INNERC + d];
            vts[bufi][d * PLD + row] = val;
        }
    };

    float mrow[8], lrow[8];
    #pragma unroll
    for (int r = 0; r < 8; ++r) { mrow[r] = -3.0e38f; lrow[r] = 0.f; }
    v8f oacc[6] = {};
    __bf16* pw = &ps[w * 16 * PLD];

    const int ntiles = (NKC + 31) / 32;
    load_tile(0, 0);
    __syncthreads();
    for (int kt = 0; kt < ntiles; ++kt) {
        int bufi = kt & 1;
        int kbase = kt * 32;
        if (kt + 1 < ntiles) load_tile(kt + 1, bufi ^ 1);

        // S = q @ k^T  (each wave: 16 rows x 32 keys)
        v8f s[2] = {};
        #pragma unroll
        for (int kk = 0; kk < 3; ++kk) {
            v16bf bk0 = load_b_frag(&ks[bufi][0 * QLD + kk * 32], QLD, lane);
            v16bf bk1 = load_b_frag(&ks[bufi][16 * QLD + kk * 32], QLD, lane);
            s[0] = wmma_bf16(qfrag[kk], bk0, s[0]);
            s[1] = wmma_bf16(qfrag[kk], bk1, s[1]);
        }
        // scale + mask invalid keys (col is per-lane uniform across VGPRs)
        #pragma unroll
        for (int j = 0; j < 2; ++j) {
            bool valid = (kbase + j * 16 + cl) < NKC;
            #pragma unroll
            for (int r = 0; r < 8; ++r)
                s[j][r] = valid ? s[j][r] * QSCALE : -3.0e38f;
        }
        // online softmax (rows r / r+8 by lane half)
        float alpha[8];
        #pragma unroll
        for (int r = 0; r < 8; ++r) {
            float t = fmaxf(s[0][r], s[1][r]);
            t = xmax16(t);
            float mnew = fmaxf(mrow[r], t);
            alpha[r] = __expf(mrow[r] - mnew);
            mrow[r] = mnew;
            float p0 = __expf(s[0][r] - mnew);
            float p1 = __expf(s[1][r] - mnew);
            s[0][r] = p0; s[1][r] = p1;
            lrow[r] = lrow[r] * alpha[r] + xsum16(p0 + p1);
        }
        // P -> per-wave LDS (C layout -> row-major), rescale O
        #pragma unroll
        for (int r = 0; r < 8; ++r) {
            int pr = r + half * 8;
            pw[pr * PLD + cl] = (__bf16)s[0][r];
            pw[pr * PLD + 16 + cl] = (__bf16)s[1][r];
        }
        #pragma unroll
        for (int t2 = 0; t2 < 6; ++t2)
            #pragma unroll
            for (int r = 0; r < 8; ++r) oacc[t2][r] *= alpha[r];
        // O += P @ V
        v16bf pa = load_a_frag(pw, PLD, lane);
        #pragma unroll
        for (int t2 = 0; t2 < 6; ++t2) {
            v16bf bv = load_b_frag(&vts[bufi][(t2 * 16) * PLD], PLD, lane);
            oacc[t2] = wmma_bf16(pa, bv, oacc[t2]);
        }
        __syncthreads();
    }
    // write O / l
    __bf16* ob = o + ((size_t)b * N2C * HEADSC + h) * DHC;
    #pragma unroll
    for (int t2 = 0; t2 < 6; ++t2)
        #pragma unroll
        for (int r = 0; r < 8; ++r) {
            int j = m0 + r + half * 8;
            int d = t2 * 16 + cl;
            ob[(size_t)j * INNERC + d] = (__bf16)(oacc[t2][r] / lrow[r]);
        }
}

// ---------------------------------------------------------------------------
// host
// ---------------------------------------------------------------------------
extern "C" void kernel_launch(void* const* d_in, const int* in_sizes, int n_in,
                              void* d_out, int out_size, void* d_ws, size_t ws_size,
                              hipStream_t stream) {
    (void)in_sizes; (void)n_in; (void)out_size; (void)ws_size;
    const float* x        = (const float*)d_in[0];
    const float* text_emb = (const float*)d_in[1];
    const float* latents  = (const float*)d_in[2];
    const float* txt_w1   = (const float*)d_in[3];
    const float* txt_b1   = (const float*)d_in[4];
    const float* txt_w2   = (const float*)d_in[5];
    const float* txt_b2   = (const float*)d_in[6];
    const float* nm_w     = (const float*)d_in[7];
    const float* nm_b     = (const float*)d_in[8];
    const float* nl_w     = (const float*)d_in[9];
    const float* nl_b     = (const float*)d_in[10];
    const float* Wq       = (const float*)d_in[11];
    const float* Wkv      = (const float*)d_in[12];
    const float* Wo       = (const float*)d_in[13];
    const float* ffln_w   = (const float*)d_in[14];
    const float* ffln_b   = (const float*)d_in[15];
    const float* ff_w1    = (const float*)d_in[16];
    const float* ff_w2    = (const float*)d_in[17];
    const float* oln_w    = (const float*)d_in[18];
    const float* oln_b    = (const float*)d_in[19];
    float* out = (float*)d_out;

    char* ws = (char*)d_ws;
    size_t off = 0;
    auto alloc = [&](size_t bytes) -> void* {
        void* p = ws + off;
        off = (off + bytes + 255) & ~(size_t)255;
        return p;
    };
    __bf16* wq_t  = (__bf16*)alloc((size_t)DEPTHC * DIMC * INNERC * 2);
    __bf16* wkv_t = (__bf16*)alloc((size_t)DEPTHC * DIMC * 2 * INNERC * 2);
    __bf16* wo_t  = (__bf16*)alloc((size_t)DEPTHC * INNERC * DIMC * 2);
    __bf16* ff1_t = (__bf16*)alloc((size_t)DEPTHC * DIMC * FFIC * 2);
    __bf16* ff2_t = (__bf16*)alloc((size_t)DEPTHC * FFIC * DIMC * 2);
    __bf16* tw1_t = (__bf16*)alloc((size_t)DIMLLM * DIMC * 2);
    __bf16* tw2_t = (__bf16*)alloc((size_t)DIMC * DIMC * 2);
    __bf16* text_bf = (__bf16*)alloc((size_t)BATCHC * NTXTC * DIMLLM * 2);
    __bf16* txt_h   = (__bf16*)alloc((size_t)BATCHC * NTXTC * DIMC * 2);
    __bf16* x_bf    = (__bf16*)alloc((size_t)BATCHC * VTOK * DIMC * 2);
    float2* xstats  = (float2*)alloc((size_t)BATCHC * VTOK * sizeof(float2));
    float*  lat     = (float*)alloc((size_t)BATCHC * N2C * DIMC * 4);
    __bf16* lm_bf   = (__bf16*)alloc((size_t)BATCHC * N2C * DIMC * 2);
    __bf16* ffin_bf = (__bf16*)alloc((size_t)BATCHC * N2C * DIMC * 2);
    __bf16* q_bf    = (__bf16*)alloc((size_t)BATCHC * N2C * INNERC * 2);
    __bf16* kv_bf   = (__bf16*)alloc((size_t)BATCHC * NKC * 2 * INNERC * 2);
    __bf16* attno   = (__bf16*)alloc((size_t)BATCHC * N2C * INNERC * 2);
    __bf16* ffh     = (__bf16*)alloc((size_t)BATCHC * N2C * FFIC * 2);

    const int BIG = 1 << 30;
    dim3 tb(32, 8);

    // weight transpose+convert (K rows, N cols -> N-major bf16)
    transpose_bf16_kernel<<<dim3(INNERC/32, DIMC/32, DEPTHC), tb, 0, stream>>>(Wq,  wq_t,  DIMC,   INNERC);
    transpose_bf16_kernel<<<dim3((2*INNERC)/32, DIMC/32, DEPTHC), tb, 0, stream>>>(Wkv, wkv_t, DIMC, 2*INNERC);
    transpose_bf16_kernel<<<dim3(DIMC/32, INNERC/32, DEPTHC), tb, 0, stream>>>(Wo,  wo_t,  INNERC, DIMC);
    transpose_bf16_kernel<<<dim3(FFIC/32, DIMC/32, DEPTHC), tb, 0, stream>>>(ff_w1, ff1_t, DIMC, FFIC);
    transpose_bf16_kernel<<<dim3(DIMC/32, FFIC/32, DEPTHC), tb, 0, stream>>>(ff_w2, ff2_t, FFIC, DIMC);
    transpose_bf16_kernel<<<dim3(DIMC/32, DIMLLM/32, 1), tb, 0, stream>>>(txt_w1, tw1_t, DIMLLM, DIMC);
    transpose_bf16_kernel<<<dim3(DIMC/32, DIMC/32, 1), tb, 0, stream>>>(txt_w2, tw2_t, DIMC, DIMC);

    size_t ntext = (size_t)BATCHC * NTXTC * DIMLLM;
    convert_bf16_kernel<<<(unsigned)((ntext + 255) / 256), 256, 0, stream>>>(text_emb, text_bf, ntext);
    row_stats_kernel<<<BATCHC * VTOK, 256, 0, stream>>>(x, x_bf, xstats, DIMC);
    init_lat_kernel<<<(BATCHC * NLATC * DIMC + 255) / 256, 256, 0, stream>>>(latents, lat);

    // txt MLP: gelu(text @ w1 + b1) @ w2 + b2 -> lat[:, 64:128]
    {
        int M = BATCHC * NTXTC;
        gemm_kernel<false, true, true, false, true><<<dim3(DIMC/64, (M+127)/128), 256, 0, stream>>>(
            text_bf, nullptr, nullptr, nullptr, tw1_t, txt_b1, nullptr, txt_h,
            M, DIMC, DIMLLM, BIG, 0, 0);
        gemm_kernel<false, true, false, false, false><<<dim3(DIMC/64, (M+127)/128), 256, 0, stream>>>(
            txt_h, nullptr, nullptr, nullptr, tw2_t, txt_b2, nullptr, lat,
            M, DIMC, DIMC, NTXTC, N2C, NLATC);
    }

    int Ml = BATCHC * N2C;      // 4096 latent rows
    int Mx = BATCHC * VTOK;     // 23328 x rows
    for (int i = 0; i < DEPTHC; ++i) {
        const __bf16* wq_i  = wq_t  + (size_t)i * DIMC * INNERC;
        const __bf16* wkv_i = wkv_t + (size_t)i * DIMC * 2 * INNERC;
        const __bf16* wo_i  = wo_t  + (size_t)i * INNERC * DIMC;
        const __bf16* f1_i  = ff1_t + (size_t)i * DIMC * FFIC;
        const __bf16* f2_i  = ff2_t + (size_t)i * FFIC * DIMC;

        // lm = LN(lat)
        ln_kernel<false><<<Ml, 256, 0, stream>>>(lat, nl_w + i*DIMC, nl_b + i*DIMC,
                                                 lm_bf, DIMC, BIG, 0, 0);
        // q = lm @ Wq
        gemm_kernel<false, false, false, false, true><<<dim3(INNERC/64, Ml/128), 256, 0, stream>>>(
            lm_bf, nullptr, nullptr, nullptr, wq_i, nullptr, nullptr, q_bf,
            Ml, INNERC, DIMC, BIG, 0, 0);
        // kv (x part, LN fused):  rows b*729+j -> b*857+j
        gemm_kernel<true, false, false, false, true><<<dim3((2*INNERC)/64, (Mx+127)/128), 256, 0, stream>>>(
            x_bf, xstats, nm_w + i*DIMC, nm_b + i*DIMC, wkv_i, nullptr, nullptr, kv_bf,
            Mx, 2*INNERC, DIMC, VTOK, NKC, 0);
        // kv (lm part): rows b*128+j -> b*857+729+j
        gemm_kernel<false, false, false, false, true><<<dim3((2*INNERC)/64, Ml/128), 256, 0, stream>>>(
            lm_bf, nullptr, nullptr, nullptr, wkv_i, nullptr, nullptr, kv_bf,
            Ml, 2*INNERC, DIMC, N2C, NKC, VTOK);
        // flash attention
        attn_kernel<<<BATCHC * HEADSC, 256, 0, stream>>>(q_bf, kv_bf, attno);
        // lat += attno @ Wo
        gemm_kernel<false, false, false, true, false><<<dim3(DIMC/64, Ml/128), 256, 0, stream>>>(
            attno, nullptr, nullptr, nullptr, wo_i, nullptr, lat, lat,
            Ml, DIMC, INNERC, BIG, 0, 0);
        // ff
        ln_kernel<false><<<Ml, 256, 0, stream>>>(lat, ffln_w + i*DIMC, ffln_b + i*DIMC,
                                                 ffin_bf, DIMC, BIG, 0, 0);
        gemm_kernel<false, false, true, false, true><<<dim3(FFIC/64, Ml/128), 256, 0, stream>>>(
            ffin_bf, nullptr, nullptr, nullptr, f1_i, nullptr, nullptr, ffh,
            Ml, FFIC, DIMC, BIG, 0, 0);
        gemm_kernel<false, false, false, true, false><<<dim3(DIMC/64, Ml/128), 256, 0, stream>>>(
            ffh, nullptr, nullptr, nullptr, f2_i, nullptr, lat, lat,
            Ml, DIMC, FFIC, BIG, 0, 0);
    }

    // out = LN(lat[:, :64]) -> f32
    ln_kernel<true><<<BATCHC * NLATC, 256, 0, stream>>>(
        lat, oln_w, oln_b, out, DIMC, NLATC, N2C, 0);
}
